// Head_9698036154792
// MI455X (gfx1250) — matrix operand order repaired and checked by reading, not compiled, not run on hardware
//
#include <hip/hip_runtime.h>
#include <hip/hip_bf16.h>

typedef __attribute__((ext_vector_type(16))) _Float16 v16h;
typedef __attribute__((ext_vector_type(8)))  _Float16 v8h;
typedef __attribute__((ext_vector_type(8)))  float    v8f;
typedef __attribute__((ext_vector_type(4)))  float    v4f;

#define BB 16
#define TT 4096
#define CC 32
#define HH 64
#define PLW 40   // padded LDS row (halves): 80B stride -> conflict-free 16-row reads

__device__ __forceinline__ v16h cat8(v8h lo, v8h hi) {
    return __builtin_shufflevector(lo, hi, 0,1,2,3,4,5,6,7,8,9,10,11,12,13,14,15);
}

// A-fragment gather for 16-bit 16x32 A: this lane's halves are [p..p+7] and [p+8..p+15]
__device__ __forceinline__ v16h load_a16(const _Float16* p) {
    v8h lo = *(const v8h*)(p);
    v8h hi = *(const v8h*)(p + 8);
    return cat8(lo, hi);
}

// ---------------------------------------------------------------------------
// Kernel 0: one-shot weight prep. Wq/Wk/Wv [C][H] f32 -> Wt [3][H][C] f16
// (transposed so proj B-fragments are contiguous v16h loads).
// ---------------------------------------------------------------------------
__global__ __launch_bounds__(256) void prep_weights(
    const float* __restrict__ Wq, const float* __restrict__ Wk,
    const float* __restrict__ Wv, _Float16* __restrict__ wt)
{
    const float* Ws[3] = { Wq, Wk, Wv };
    for (int idx = threadIdx.x; idx < 3 * HH * CC; idx += 256) {
        const int w = idx / (HH * CC);
        const int r = idx % (HH * CC);
        const int h = r / CC;
        const int c = r % CC;
        wt[idx] = (_Float16)Ws[w][c * HH + h];
    }
}

// ---------------------------------------------------------------------------
// Kernel 1: q = x@Wq, k = x@Wk, v = x@Wv   (f32 in -> f16 out)
// q,k stored row-major [B][T][H]; v stored transposed [B][H][T].
// 4 independent waves per block (share weights via WGP$); 12 WMMAs/wave.
// ---------------------------------------------------------------------------
__global__ __launch_bounds__(128) void proj_kernel(
    const float* __restrict__ x, const _Float16* __restrict__ wt,
    _Float16* __restrict__ qh, _Float16* __restrict__ kh, _Float16* __restrict__ vt)
{
    const int blk  = blockIdx.x;
    const int wave = threadIdx.x >> 5;
    const int b    = blk >> 6;                       // 64 blocks x 4 waves = 256 tiles
    const int rt   = ((blk & 63) << 2) + wave;
    const int lane = threadIdx.x & 31;
    const int n16  = lane & 15;
    const int hlf  = lane >> 4;

    // A fragment: x tile [16 x 32] fp32 -> f16 (vector loads + pk converts)
    const float* xr = x + ((size_t)b * TT + rt * 16 + n16) * CC + (hlf ? 16 : 0);
    v4f x0 = *(const v4f*)(xr);
    v4f x1 = *(const v4f*)(xr + 4);
    v4f x2 = *(const v4f*)(xr + 8);
    v4f x3 = *(const v4f*)(xr + 12);
    v16h a;
#pragma unroll
    for (int e = 0; e < 4; ++e) {
        a[e]      = (_Float16)x0[e];
        a[e + 4]  = (_Float16)x1[e];
        a[e + 8]  = (_Float16)x2[e];
        a[e + 12] = (_Float16)x3[e];
    }

#pragma unroll
    for (int w = 0; w < 3; ++w) {
#pragma unroll
        for (int t = 0; t < 4; ++t) {           // H = 64 -> 4 N tiles
            const int col = t * 16 + n16;
            // B fragment: contiguous 16 halves of Wt[w][col][hlf*16 ..]
            v16h bf = *(const v16h*)(wt + ((size_t)w * HH + col) * CC + hlf * 16);
            v8f c = {};
            c = __builtin_amdgcn_wmma_f32_16x16x32_f16(false, a, false, bf,
                                                       (short)0, c, false, false);
#pragma unroll
            for (int j = 0; j < 8; ++j) {
                const int grow = rt * 16 + j + hlf * 8;   // C-layout row
                const _Float16 val = (_Float16)c[j];
                if (w == 0)      qh[((size_t)b * TT + grow) * HH + col] = val;
                else if (w == 1) kh[((size_t)b * TT + grow) * HH + col] = val;
                else             vt[((size_t)b * HH + col) * TT + grow] = val;
            }
        }
    }
}

// ---------------------------------------------------------------------------
// One 32-key flash-attention step. MASK=true only for diagonal tiles.
// No hardware barrier: LDS ops are in-order within a wave (per-wave pl slice);
// only a compiler reordering fence is required between P write and re-read.
// ---------------------------------------------------------------------------
template<bool MASK>
__device__ __forceinline__ void attn_step(
    int b, int qt, int kt, int n16, int hlf, int ktmax,
    const _Float16* __restrict__ kh, const _Float16* __restrict__ vt,
    const v16h& qa0, const v16h& qa1,
    float (&m_i)[8], float (&l_i)[8], v8f (&o)[4],
    _Float16 (*pl)[PLW])
{
    const size_t kbase = ((size_t)b * TT + kt * 32 + n16) * HH + hlf * 16;
    // prefetch next key tile's K lines (L2-resident; pulls into WGP level)
    if (kt < ktmax) {
        __builtin_prefetch(kh + kbase + 32 * HH, 0, 3);
        __builtin_prefetch(kh + kbase + 48 * HH, 0, 3);
    }

    // ---- S = Q K^T : two 16-key sub-tiles, two H chunks each
    v8f s0 = {}, s1 = {};
    {
        const _Float16* krow = kh + kbase;
        v16h k0 = *(const v16h*)(krow);
        v16h k1 = *(const v16h*)(krow + 32);
        s0 = __builtin_amdgcn_wmma_f32_16x16x32_f16(false, qa0, false, k0,
                                                    (short)0, s0, false, false);
        s0 = __builtin_amdgcn_wmma_f32_16x16x32_f16(false, qa1, false, k1,
                                                    (short)0, s0, false, false);
    }
    {
        const _Float16* krow = kh + kbase + 16 * HH;
        v16h k0 = *(const v16h*)(krow);
        v16h k1 = *(const v16h*)(krow + 32);
        s1 = __builtin_amdgcn_wmma_f32_16x16x32_f16(false, qa0, false, k0,
                                                    (short)0, s1, false, false);
        s1 = __builtin_amdgcn_wmma_f32_16x16x32_f16(false, qa1, false, k1,
                                                    (short)0, s1, false, false);
    }

    // ---- scale (+ causal mask on diagonal tiles only)
    float p0[8], p1[8];
    const int gq_base = qt * 16 + hlf * 8;
    const int gk0 = kt * 32 + n16;
#pragma unroll
    for (int j = 0; j < 8; ++j) {
        float v0 = s0[j] * 0.125f;
        float v1 = s1[j] * 0.125f;
        if (MASK) {
            const int gq = gq_base + j;
            v0 = (gk0      <= gq) ? v0 : -__builtin_inff();
            v1 = (gk0 + 16 <= gq) ? v1 : -__builtin_inff();
        }
        p0[j] = v0; p1[j] = v1;
    }

    // ---- online softmax, per row (16 lanes of a half share a row)
#pragma unroll
    for (int j = 0; j < 8; ++j) {
        float mt = fmaxf(p0[j], p1[j]);
#pragma unroll
        for (int d = 1; d < 16; d <<= 1)
            mt = fmaxf(mt, __shfl_xor(mt, d, 32));
        const float mn = fmaxf(m_i[j], mt);
        const float a0 = __expf(p0[j] - mn);
        const float a1 = __expf(p1[j] - mn);
        float rs = a0 + a1;
#pragma unroll
        for (int d = 1; d < 16; d <<= 1)
            rs += __shfl_xor(rs, d, 32);
        const float alpha = __expf(m_i[j] - mn);
        m_i[j] = mn;
        l_i[j] = l_i[j] * alpha + rs;
#pragma unroll
        for (int t = 0; t < 4; ++t) o[t][j] *= alpha;
        const int M = j + hlf * 8;
        pl[M][n16]      = (_Float16)a0;
        pl[M][16 + n16] = (_Float16)a1;
    }
    asm volatile("" ::: "memory");   // compiler fence: keep LDS read after writes
                                     // (hardware keeps per-wave LDS ops in order)

    // ---- reload P as A-fragment (16x32, K = 32 keys)
    const int off = hlf ? 16 : 0;
    v16h pa = load_a16(&pl[n16][off]);

    // ---- O += P @ V  (V^T stored [H][T] -> contiguous B-fragment loads)
    const size_t vbase = (size_t)b * HH * TT + n16 * TT + kt * 32 + hlf * 16;
#pragma unroll
    for (int t = 0; t < 4; ++t) {
        const _Float16* vrow = vt + vbase + (size_t)t * 16 * TT;
        if (kt < ktmax) __builtin_prefetch(vrow + 32, 0, 3);
        v16h vb = *(const v16h*)vrow;
        o[t] = __builtin_amdgcn_wmma_f32_16x16x32_f16(false, pa, false, vb,
                                                      (short)0, o[t], false, false);
    }
}

// ---------------------------------------------------------------------------
// Kernel 2: fused causal flash attention.
// 4 waves per block = 4 consecutive query tiles on one WGP: their kt loops
// stay within +-1 of each other, so K/V tiles are shared through the WGP$
// (~4x less L2 traffic). Waves are fully independent (no barriers).
// ---------------------------------------------------------------------------
__global__ __launch_bounds__(128) void attn_kernel(
    const _Float16* __restrict__ qh, const _Float16* __restrict__ kh,
    const _Float16* __restrict__ vt, float* __restrict__ out)
{
    __shared__ _Float16 pl[4][16][PLW];   // per-wave prob-tile bounce slice

    const int blk  = blockIdx.x;
    const int wave = threadIdx.x >> 5;
    const int b    = blk >> 6;                       // 64 blocks x 4 waves = 256 tiles
    const int qt   = ((blk & 63) << 2) + wave;
    const int lane = threadIdx.x & 31;
    const int n16  = lane & 15;
    const int hlf  = lane >> 4;

    // Q A-fragments, 2 chunks over H=64
    const _Float16* qrow = qh + ((size_t)b * TT + qt * 16 + n16) * HH;
    const int aoff = hlf ? 16 : 0;
    v16h qa0 = load_a16(qrow + aoff);
    v16h qa1 = load_a16(qrow + 32 + aoff);

    float m_i[8], l_i[8];
    v8f o[4];
#pragma unroll
    for (int j = 0; j < 8; ++j) { m_i[j] = -__builtin_inff(); l_i[j] = 0.f; }
#pragma unroll
    for (int t = 0; t < 4; ++t)
#pragma unroll
        for (int e = 0; e < 8; ++e) o[t][e] = 0.f;

    const int ktmax = (qt * 16 + 15) >> 5;              // last tile touching causal region
    const int nfull = (qt >= 2) ? ((16 * qt - 31) / 32 + 1) : 0;  // fully-unmasked tiles

    int kt = 0;
    for (; kt < nfull; ++kt)
        attn_step<false>(b, qt, kt, n16, hlf, ktmax, kh, vt, qa0, qa1, m_i, l_i, o, pl[wave]);
    for (; kt <= ktmax; ++kt)
        attn_step<true >(b, qt, kt, n16, hlf, ktmax, kh, vt, qa0, qa1, m_i, l_i, o, pl[wave]);

    // ---- epilogue: out = O / l  (fp32, [B][T][H])
#pragma unroll
    for (int j = 0; j < 8; ++j) {
        const float inv = 1.0f / l_i[j];
        const size_t rbase = ((size_t)b * TT + qt * 16 + j + hlf * 8) * HH + n16;
#pragma unroll
        for (int t = 0; t < 4; ++t)
            out[rbase + t * 16] = o[t][j] * inv;
    }
}

extern "C" void kernel_launch(void* const* d_in, const int* in_sizes, int n_in,
                              void* d_out, int out_size, void* d_ws, size_t ws_size,
                              hipStream_t stream) {
    const float* x  = (const float*)d_in[0];
    const float* Wk = (const float*)d_in[1];
    const float* Wq = (const float*)d_in[2];
    const float* Wv = (const float*)d_in[3];
    float* out = (float*)d_out;

    const size_t elems = (size_t)BB * TT * HH;      // 4 Mi f16 elements each
    _Float16* qh = (_Float16*)d_ws;
    _Float16* kh = qh + elems;
    _Float16* vt = kh + elems;
    _Float16* wt = vt + elems;                      // 3*64*32 f16 = 12 KB extra

    prep_weights<<<1, 256, 0, stream>>>(Wq, Wk, Wv, wt);

    const int nblocks = BB * (TT / 16) / 4;         // 1024 four-wave blocks
    proj_kernel<<<nblocks, 128, 0, stream>>>(x, wt, qh, kh, vt);
    attn_kernel<<<nblocks, 128, 0, stream>>>(qh, kh, vt, out);
}
